// Retention_12833362280748
// MI455X (gfx1250) — matrix operand order repaired and checked by reading, not compiled, not run on hardware
//
#include <hip/hip_runtime.h>

// ---------------------------------------------------------------------------
// Types for CDNA5 WMMA (wave32): v_wmma_f32_16x16x32_bf16
// ---------------------------------------------------------------------------
typedef __bf16 bf16;
typedef __attribute__((ext_vector_type(16))) __bf16 v16bf;
typedef __attribute__((ext_vector_type(8)))  float  v8f;

__device__ __forceinline__ v8f v8f_zero() {
    v8f z;
#pragma unroll
    for (int i = 0; i < 8; ++i) z[i] = 0.0f;
    return z;
}

__device__ __forceinline__ v8f wmma_bf16(v16bf a, v16bf b, v8f c) {
    // (neg_a, A, neg_b, B, c_mod, C, reuse_a, reuse_b)
    return __builtin_amdgcn_wmma_f32_16x16x32_bf16(false, a, false, b,
                                                   (short)0, c, false, false);
}

// Problem constants
#define BSZ   2
#define LSEQ  4096
#define EDIM  1024
#define MROWS (BSZ * LSEQ)      // 8192
#define HALF  (EDIM / 2)        // 512
#define LOG2_GAMMA  (-0.04580368961312478f)   // log2(0.96875)
#define LOG2_THETA  (13.287712379549449f)     // log2(10000)

// ---------------------------------------------------------------------------
// Kernel 1: Y[m,n] = sum_k X[m,k] * W[n,k], with fused xPos epilogue.
//   mode 0: plain (V projection), mode 1: xPos forward (Q), mode 2: xPos inverse (K)
// Output written as bf16 (f32 accumulate inside WMMA).
// Block: 256 threads = 8 waves (4 along M x 2 along N). Tile 128x64, K-step 32.
// ---------------------------------------------------------------------------
#define MBLK 128
#define NBLK 64
#define KBLK 32
#define LDA  (KBLK + 8)   // padded LDS row stride (bf16 elems)
#define LDB  (KBLK + 8)

__global__ __launch_bounds__(256, 2)
void proj_xpos_kernel(const float* __restrict__ X,   // [MROWS, EDIM]
                      const float* __restrict__ W,   // [EDIM, EDIM] (row n = out feature)
                      bf16* __restrict__ Y,          // [MROWS, EDIM] bf16
                      int mode)
{
    __shared__ bf16 As[MBLK * LDA];
    __shared__ bf16 Bs[NBLK * LDB];

    const int tid  = threadIdx.x;
    const int lane = tid & 31;
    const int w    = tid >> 5;
    const int wm   = w & 3;     // 4 waves along M
    const int wn   = w >> 2;    // 2 waves along N
    const int bm   = blockIdx.x * MBLK;
    const int bn   = blockIdx.y * NBLK;

    v8f acc[2][2];
#pragma unroll
    for (int mt = 0; mt < 2; ++mt)
#pragma unroll
        for (int nt = 0; nt < 2; ++nt) acc[mt][nt] = v8f_zero();

    // Fragment lane mappings (ISA 7.12.2, 16-bit A 16x32 / B 32x16)
    const int arow = lane & 15;
    const int akb  = (lane < 16) ? 0 : 8;
    const int bcol = lane & 15;
    const int bkb  = (lane < 16) ? 0 : 8;

    for (int k0 = 0; k0 < EDIM; k0 += KBLK) {
        // Stage A tile (128x32) f32 -> bf16: thread t covers half a row
        {
            const int r = tid >> 1;
            const int c = (tid & 1) * 16;
            const float* src = X + (size_t)(bm + r) * EDIM + k0 + c;
            bf16* dst = As + r * LDA + c;
#pragma unroll
            for (int j = 0; j < 16; ++j) dst[j] = (bf16)src[j];
        }
        // Stage B tile (64x32)
        {
            const int r = tid >> 2;
            const int c = (tid & 3) * 8;
            const float* src = W + (size_t)(bn + r) * EDIM + k0 + c;
            bf16* dst = Bs + r * LDB + c;
#pragma unroll
            for (int j = 0; j < 8; ++j) dst[j] = (bf16)src[j];
        }
        __syncthreads();

        v16bf afrag[2], bfrag[2];
#pragma unroll
        for (int mt = 0; mt < 2; ++mt) {
            const bf16* ap = As + (wm * 32 + mt * 16 + arow) * LDA + akb;
#pragma unroll
            for (int e = 0; e < 16; ++e) afrag[mt][e] = ap[(e < 8) ? e : e + 8];
        }
#pragma unroll
        for (int nt = 0; nt < 2; ++nt) {
            const bf16* bp = Bs + (wn * 32 + nt * 16 + bcol) * LDB + bkb;
#pragma unroll
            for (int e = 0; e < 16; ++e) bfrag[nt][e] = bp[(e < 8) ? e : e + 8];
        }
#pragma unroll
        for (int mt = 0; mt < 2; ++mt)
#pragma unroll
            for (int nt = 0; nt < 2; ++nt)
                acc[mt][nt] = wmma_bf16(afrag[mt], bfrag[nt], acc[mt][nt]);
        __syncthreads();
    }

    // ---- Epilogue: fused xPos.
    // Per-lane column is fixed per N-tile, so all column-only transcendentals
    // (log2 of the scale base, the rotary frequency) hoist out of the row loop:
    // scale = exp2(log2(base) * pos/512 * sgn), freq = exp2(-i/512 * log2(1e4)).
    // rotate_half partner lives in lane^1 -> one shfl_xor.
    const float inv14E = 1.0f / (1.4f * (float)EDIM);
    const float sgn    = (mode == 2) ? -1.0f : 1.0f;  // inverse scale = negated exponent
    const int   rowoff = (lane >> 4) * 8;

    float l2base[2], freqc[2];
#pragma unroll
    for (int nt = 0; nt < 2; ++nt) {
        const int col = bn + wn * 32 + nt * 16 + (lane & 15);
        const int i   = col >> 1;
        l2base[nt] = __log2f((2.0f * (float)i + 0.4f * (float)EDIM) * inv14E);
        freqc[nt]  = exp2f(-(float)i * (LOG2_THETA / (float)HALF));
    }

#pragma unroll
    for (int nt = 0; nt < 2; ++nt) {
        const int col = bn + wn * 32 + nt * 16 + (lane & 15);
#pragma unroll
        for (int mt = 0; mt < 2; ++mt) {
#pragma unroll
            for (int r = 0; r < 8; ++r) {
                const int m = bm + wm * 32 + mt * 16 + rowoff + r;
                float x  = acc[mt][nt][r];
                float xn = __shfl_xor(x, 1, 32);   // uniform control flow
                float y;
                if (mode == 0) {
                    y = x;
                } else {
                    const float pos = (float)(m & (LSEQ - 1));
                    const float sc  = exp2f(l2base[nt] * pos * (sgn / 512.0f));
                    float s, c;
                    __sincosf(pos * freqc[nt], &s, &c);
                    const float rot = (col & 1) ? xn : -xn;
                    y = (x * c + rot * s) * sc;
                }
                Y[(size_t)m * EDIM + col] = (bf16)y;
            }
        }
    }
}

// ---------------------------------------------------------------------------
// Kernel 2: fused retention  O[q,:] = sum_{k<=q} gamma^(q-k) (Q[q]·K[k]) V[k,:]
// Block = (batch, 32-query tile), 256 threads = 8 waves.
//   Phase A: waves split the E=1024 reduction of S = Q·K^T (16x wmma each),
//            partials to LDS.
//   Phase C: reduce partials, apply decay mask, requantize S to bf16 in LDS.
//   Phase D: each wave owns a 128-col slice of S·V (16x wmma), V staged
//            transposed in LDS so B-fragments are contiguous ds reads.
// Causal chunk skipping: loop runs only kc <= q0.  K/V chunks (16 MB/batch
// bf16) stay resident in the 192 MB L2, so cross-block re-reads are cheap.
// ---------------------------------------------------------------------------
#define KC     32
#define VT_LD  34            // padded k-stride of Vt rows

__global__ __launch_bounds__(256, 1)
void retention_kernel(const bf16* __restrict__ Q,   // [MROWS, EDIM] (xPos'ed)
                      const bf16* __restrict__ K,   // [MROWS, EDIM] (xPos inv)
                      const bf16* __restrict__ V,   // [MROWS, EDIM]
                      float* __restrict__ O)        // [MROWS, EDIM] f32
{
    __shared__ float Spart[8][32][32];       // 32 KB  per-wave partial S
    __shared__ bf16  Sbf[32][33];            //  2 KB  masked S, bf16
    __shared__ bf16  Vt[EDIM * VT_LD];       // 68 KB  V chunk transposed [n][k]

    const int tid  = threadIdx.x;
    const int lane = tid & 31;
    const int w    = tid >> 5;
    const int q0   = blockIdx.x * 32;
    const size_t rowbase = (size_t)blockIdx.y * LSEQ;

    const int arow = lane & 15;
    const int akb  = (lane < 16) ? 0 : 8;
    const int bcol = lane & 15;
    const int bkb  = (lane < 16) ? 0 : 8;
    const int rowoff = (lane >> 4) * 8;

    v8f oacc[2][8];
#pragma unroll
    for (int mt = 0; mt < 2; ++mt)
#pragma unroll
        for (int nt = 0; nt < 8; ++nt) oacc[mt][nt] = v8f_zero();

    for (int kc = 0; kc <= q0; kc += KC) {
        // Hint the V chunk toward the caches (global_prefetch_b8)
        {
            const int pr = tid >> 3;
            const int pc = (tid & 7) * 128;
            __builtin_prefetch(V + (rowbase + kc + pr) * EDIM + pc, 0, 1);
        }

        // ---- Phase A: partial S over this wave's e-slice [w*128, w*128+128)
        v8f sacc[2][2];
#pragma unroll
        for (int mt = 0; mt < 2; ++mt)
#pragma unroll
            for (int nt = 0; nt < 2; ++nt) sacc[mt][nt] = v8f_zero();

        for (int es = 0; es < 4; ++es) {
            const int e0 = w * 128 + es * 32;
            v16bf aq[2], bk[2];
#pragma unroll
            for (int mt = 0; mt < 2; ++mt) {
                const bf16* ap = Q + (rowbase + q0 + mt * 16 + arow) * EDIM + e0 + akb;
#pragma unroll
                for (int e = 0; e < 16; ++e) aq[mt][e] = ap[(e < 8) ? e : e + 8];
            }
#pragma unroll
            for (int nt = 0; nt < 2; ++nt) {
                const bf16* bp = K + (rowbase + kc + nt * 16 + bcol) * EDIM + e0 + bkb;
#pragma unroll
                for (int e = 0; e < 16; ++e) bk[nt][e] = bp[(e < 8) ? e : e + 8];
            }
#pragma unroll
            for (int mt = 0; mt < 2; ++mt)
#pragma unroll
                for (int nt = 0; nt < 2; ++nt)
                    sacc[mt][nt] = wmma_bf16(aq[mt], bk[nt], sacc[mt][nt]);
        }
#pragma unroll
        for (int mt = 0; mt < 2; ++mt)
#pragma unroll
            for (int nt = 0; nt < 2; ++nt)
#pragma unroll
                for (int r = 0; r < 8; ++r)
                    Spart[w][mt * 16 + rowoff + r][nt * 16 + (lane & 15)] = sacc[mt][nt][r];

        // ---- Phase B: stage V chunk transposed: Vt[n][k] = V[kc+k][n]
        for (int j = tid; j < KC * (EDIM / 2); j += 256) {
            const int k  = j >> 9;            // 0..31
            const int n2 = (j & 511) * 2;     // even column
            union { unsigned int u; bf16 h[2]; } cv;
            cv.u = *(const unsigned int*)(V + (rowbase + kc + k) * EDIM + n2);
            Vt[(n2 + 0) * VT_LD + k] = cv.h[0];
            Vt[(n2 + 1) * VT_LD + k] = cv.h[1];
        }
        __syncthreads();   // Spart & Vt ready

        // ---- Phase C: reduce partials, decay mask, requantize to bf16
        for (int j = tid; j < 32 * 32; j += 256) {
            const int u = j >> 5, v = j & 31;
            float s = 0.0f;
#pragma unroll
            for (int p = 0; p < 8; ++p) s += Spart[p][u][v];
            const int dq = (q0 + u) - (kc + v);
            s = (dq >= 0) ? s * exp2f((float)dq * LOG2_GAMMA) : 0.0f;
            Sbf[u][v] = (bf16)s;
        }
        __syncthreads();   // Sbf ready

        // ---- Phase D: oacc += S(32x32) * V(32x128-slice), 16 wmma
        v16bf as[2];
#pragma unroll
        for (int mt = 0; mt < 2; ++mt)
#pragma unroll
            for (int e = 0; e < 16; ++e)
                as[mt][e] = Sbf[mt * 16 + arow][akb + ((e < 8) ? e : e + 8)];
#pragma unroll
        for (int nt = 0; nt < 8; ++nt) {
            const int n = w * 128 + nt * 16 + bcol;
            v16bf bv;
#pragma unroll
            for (int e = 0; e < 16; ++e) bv[e] = Vt[n * VT_LD + bkb + ((e < 8) ? e : e + 8)];
            oacc[0][nt] = wmma_bf16(as[0], bv, oacc[0][nt]);
            oacc[1][nt] = wmma_bf16(as[1], bv, oacc[1][nt]);
        }
        __syncthreads();   // done reading Sbf/Vt before next chunk overwrites
    }

    // ---- Store output (f32)
#pragma unroll
    for (int mt = 0; mt < 2; ++mt)
#pragma unroll
        for (int nt = 0; nt < 8; ++nt) {
            const int col = w * 128 + nt * 16 + (lane & 15);
#pragma unroll
            for (int r = 0; r < 8; ++r) {
                const int row = q0 + mt * 16 + rowoff + r;
                O[(rowbase + row) * EDIM + col] = oacc[mt][nt][r];
            }
        }
}

// ---------------------------------------------------------------------------
// Host launcher
// ---------------------------------------------------------------------------
extern "C" void kernel_launch(void* const* d_in, const int* in_sizes, int n_in,
                              void* d_out, int out_size, void* d_ws, size_t ws_size,
                              hipStream_t stream) {
    const float* query = (const float*)d_in[0];
    const float* key_  = (const float*)d_in[1];
    const float* value = (const float*)d_in[2];
    const float* Wq    = (const float*)d_in[3];
    const float* Wk    = (const float*)d_in[4];
    const float* Wv    = (const float*)d_in[5];
    float* out = (float*)d_out;

    const size_t planeElems = (size_t)MROWS * EDIM;
    bf16* qb = (bf16*)d_ws;
    bf16* kb = qb + planeElems;
    bf16* vb = kb + planeElems;

    dim3 gProj(MROWS / MBLK, EDIM / NBLK);   // 64 x 16
    proj_xpos_kernel<<<gProj, 256, 0, stream>>>(query, Wq, qb, 1);  // Q, xpos fwd
    proj_xpos_kernel<<<gProj, 256, 0, stream>>>(key_,  Wk, kb, 2);  // K, xpos inv
    proj_xpos_kernel<<<gProj, 256, 0, stream>>>(value, Wv, vb, 0);  // V, plain

    dim3 gRet(LSEQ / 32, BSZ);               // 128 x 2
    retention_kernel<<<gRet, 256, 0, stream>>>(qb, kb, vb, out);
}
